// GatedRGCN_86380382257759
// MI455X (gfx1250) — matrix operand order repaired and checked by reading, not compiled, not run on hardware
//
#include <hip/hip_runtime.h>
#include <hip/hip_bf16.h>

// ---------------------------------------------------------------------------
// Gated RGCN for gfx1250 (MI455X): f16 WMMA GEMMs (f32 accumulate) + atomic
// scatter-mean edge aggregation. wave32, 16x16x32 WMMA tiles.
// ---------------------------------------------------------------------------

typedef _Float16 v16h __attribute__((ext_vector_type(16)));
typedef _Float16 h8   __attribute__((ext_vector_type(8)));
typedef float    v8f  __attribute__((ext_vector_type(8)));

#define GF  768            // feature dim F
#define GK1 (4 * GF)       // fused RGCN K: [x | agg0 | agg1 | agg2]
#define GK2 (2 * GF)       // fused gate K: [h | qi]

// ---- WMMA fragment loaders (wave32, 16x16x32 f16) -------------------------
// A 16x32 (MxK), row-major A[M][lda]:
//   lane<16 : row=lane,    a[0..7]=K[0..7],  a[8..15]=K[16..23]
//   lane>=16: row=lane-16, a[0..7]=K[8..15], a[8..15]=K[24..31]
__device__ __forceinline__ v16h load_a_frag(const _Float16* __restrict__ A,
                                            int lda, int row0, int k0, int lane) {
  int r  = lane & 15;
  int kh = (lane >> 4) << 3;                       // 0 or 8
  const _Float16* p = A + (size_t)(row0 + r) * lda + k0 + kh;
  h8 lo = *(const h8*)(p);
  h8 hi = *(const h8*)(p + 16);
  return __builtin_shufflevector(lo, hi, 0,1,2,3,4,5,6,7,8,9,10,11,12,13,14,15);
}

// B 32x16 (KxN) from W stored row-major [N][ldw] (i.e. B^T):
//   lane<16 : col=lane,    b[0..7]=K[0..7],  b[8..15]=K[16..23]
//   lane>=16: col=lane-16, b[0..7]=K[8..15], b[8..15]=K[24..31]
__device__ __forceinline__ v16h load_b_frag(const _Float16* __restrict__ W,
                                            int ldw, int n0, int k0, int lane) {
  int n  = lane & 15;
  int ko = (lane >> 4) << 3;                       // 0 or 8
  const _Float16* p = W + (size_t)(n0 + n) * ldw + k0 + ko;
  h8 lo = *(const h8*)(p);
  h8 hi = *(const h8*)(p + 16);
  return __builtin_shufflevector(lo, hi, 0,1,2,3,4,5,6,7,8,9,10,11,12,13,14,15);
}

__device__ __forceinline__ v8f wmma_f16(v16h a, v16h b, v8f c) {
  return __builtin_amdgcn_wmma_f32_16x16x32_f16(false, a, false, b,
                                                (short)0, c, false, false);
}

// C/D 16x16 f32: VGPR v -> row v (lanes 0-15) / row v+8 (lanes 16-31), col=lane&15
__device__ __forceinline__ void store_c_tile(v8f c, float* __restrict__ C, int ldc,
                                             _Float16* __restrict__ C16, int ldc16,
                                             const float* __restrict__ bias,
                                             int relu, int m0, int n0, int lane) {
  int col = lane & 15;
  int rb  = (lane >> 4) << 3;
  float bv = bias ? bias[n0 + col] : 0.0f;
#pragma unroll
  for (int v = 0; v < 8; ++v) {
    int row = m0 + rb + v;
    float x = c[v] + bv;
    if (relu) x = fmaxf(x, 0.0f);
    C[(size_t)row * ldc + n0 + col] = x;
    if (C16) C16[(size_t)row * ldc16 + n0 + col] = (_Float16)x;
  }
}

// ---- GEMM: C[M][Nc] = A[M][K] * W[Nc][K]^T (+bias) (+relu), f16 in / f32 out.
// Block = 256 thr = 8 waves as 4(M)x2(N); wave tile = 32x32 (2x2 WMMA tiles);
// block tile = 128x64. M%128==0, Nc%64==0, K%32==0.
__global__ __launch_bounds__(256) void gemm_f16_wmma(
    const _Float16* __restrict__ A, int lda,
    const _Float16* __restrict__ W, int ldw,
    const float* __restrict__ bias,
    float* __restrict__ C, int ldc,
    _Float16* __restrict__ C16, int ldc16,
    int K, int relu) {
  int lane = threadIdx.x & 31;
  int w    = threadIdx.x >> 5;
  int wm   = w >> 1;
  int wn   = w & 1;
  int m0   = blockIdx.x * 128 + wm * 32;
  int n0   = blockIdx.y * 64  + wn * 32;

  v8f c00 = {}, c01 = {}, c10 = {}, c11 = {};
  for (int k = 0; k < K; k += 32) {
    v16h a0 = load_a_frag(A, lda, m0,      k, lane);
    v16h a1 = load_a_frag(A, lda, m0 + 16, k, lane);
    v16h b0 = load_b_frag(W, ldw, n0,      k, lane);
    v16h b1 = load_b_frag(W, ldw, n0 + 16, k, lane);
    c00 = wmma_f16(a0, b0, c00);
    c01 = wmma_f16(a0, b1, c01);
    c10 = wmma_f16(a1, b0, c10);
    c11 = wmma_f16(a1, b1, c11);
  }
  store_c_tile(c00, C, ldc, C16, ldc16, bias, relu, m0,      n0,      lane);
  store_c_tile(c01, C, ldc, C16, ldc16, bias, relu, m0,      n0 + 16, lane);
  store_c_tile(c10, C, ldc, C16, ldc16, bias, relu, m0 + 16, n0,      lane);
  store_c_tile(c11, C, ldc, C16, ldc16, bias, relu, m0 + 16, n0 + 16, lane);
}

// ---- helpers --------------------------------------------------------------
__global__ void zero_f32(float* __restrict__ p, long n) {
  long i = (long)blockIdx.x * blockDim.x + threadIdx.x;
  long stride = (long)gridDim.x * blockDim.x;
  for (; i < n; i += stride) p[i] = 0.0f;
}

// D[r][c] = (f16) S[r*srow + soff + c], c < drow
__global__ void conv_rows_f16(const float* __restrict__ S, int srow, int soff,
                              _Float16* __restrict__ D, int drow, long total) {
  long i = (long)blockIdx.x * blockDim.x + threadIdx.x;
  if (i >= total) return;
  long r = i / drow;
  int  c = (int)(i % drow);
  D[i] = (_Float16)S[(size_t)r * srow + soff + c];
}

// Wcat[o][k] : k<F -> Wroot[k][o] ; else Wrel[r][i][o]  (r=(k-F)/F, i=(k-F)%F)
__global__ void build_wcat(const float* __restrict__ Wroot,
                           const float* __restrict__ Wrel,
                           _Float16* __restrict__ Wcat, long total) {
  long idx = (long)blockIdx.x * blockDim.x + threadIdx.x;
  if (idx >= total) return;
  int o = (int)(idx / GK1);
  int k = (int)(idx % GK1);
  float v;
  if (k < GF) {
    v = Wroot[(size_t)k * GF + o];
  } else {
    int r = (k - GF) / GF, i = (k - GF) % GF;
    v = Wrel[((size_t)r * GF + i) * GF + o];
  }
  Wcat[idx] = (_Float16)v;
}

__global__ void count_edges(const int* __restrict__ dst, const int* __restrict__ et,
                            float* __restrict__ cnt, int E, int N) {
  int e = blockIdx.x * blockDim.x + threadIdx.x;
  if (e >= E) return;
  atomicAdd(&cnt[et[e] * N + dst[e]], 1.0f);
}

__global__ __launch_bounds__(256) void scatter_edges(
    const float* __restrict__ X, const int* __restrict__ src,
    const int* __restrict__ dst, const int* __restrict__ et,
    float* __restrict__ agg, int N) {
  int e = blockIdx.x;
  int s = src[e], d = dst[e], r = et[e];
  const float* xi = X + (size_t)s * GF;
  float* o = agg + ((size_t)r * N + d) * GF;
  for (int f = threadIdx.x; f < GF; f += blockDim.x) atomicAdd(&o[f], xi[f]);
}

// Acat[n][c]: c<F -> x[n][c]; else agg[r][n][f]/max(cnt,1)
__global__ void build_acat(const float* __restrict__ X, const float* __restrict__ agg,
                           const float* __restrict__ cnt, _Float16* __restrict__ Acat,
                           int N, long total) {
  long idx = (long)blockIdx.x * blockDim.x + threadIdx.x;
  if (idx >= total) return;
  int n = (int)(idx / GK1);
  int c = (int)(idx % GK1);
  float v;
  if (c < GF) {
    v = X[(size_t)n * GF + c];
  } else {
    int r = (c - GF) / GF, f = (c - GF) % GF;
    v = agg[((size_t)r * N + n) * GF + f] / fmaxf(cnt[r * N + n], 1.0f);
  }
  Acat[idx] = (_Float16)v;
}

// qi[bn][f] = sum_l sigmoid(hg[bn][f] + qg[b*L+l][f]) * q[b][l][f]
__global__ __launch_bounds__(256) void qi_kernel(
    const float* __restrict__ hg, const float* __restrict__ qg,
    const float* __restrict__ q, float* __restrict__ qi,
    _Float16* __restrict__ qi16, int qi16_ld, int Nper, int L) {
  int bn = blockIdx.x;
  int b  = bn / Nper;
  const float* hgr = hg + (size_t)bn * GF;
  const float* qgb = qg + (size_t)b * L * GF;
  const float* qb  = q  + (size_t)b * L * GF;
  for (int f = threadIdx.x; f < GF; f += blockDim.x) {
    float hv  = hgr[f];
    float acc = 0.0f;
    for (int l = 0; l < L; ++l) {
      float g  = hv + qgb[(size_t)l * GF + f];
      float sg = 1.0f / (1.0f + __expf(-g));
      acc += sg * qb[(size_t)l * GF + f];
    }
    qi[(size_t)bn * GF + f] = acc;
    qi16[(size_t)bn * qi16_ld + f] = (_Float16)acc;
  }
}

// out = sigmoid(z)*tanh(qi) + (1-sigmoid(z))*h
__global__ void combine_gate(const float* __restrict__ z, const float* __restrict__ qi,
                             const float* __restrict__ h, float* __restrict__ out, long n) {
  long i = (long)blockIdx.x * blockDim.x + threadIdx.x;
  if (i >= n) return;
  float a = 1.0f / (1.0f + __expf(-z[i]));
  float t = tanhf(qi[i]);
  out[i] = a * t + (1.0f - a) * h[i];
}

// ---------------------------------------------------------------------------
extern "C" void kernel_launch(void* const* d_in, const int* in_sizes, int n_in,
                              void* d_out, int out_size, void* d_ws, size_t ws_size,
                              hipStream_t stream) {
  const float* x    = (const float*)d_in[0];
  const int*   ei   = (const int*)d_in[1];
  const int*   et   = (const int*)d_in[2];
  const float* q    = (const float*)d_in[3];
  const float* Wr1  = (const float*)d_in[4];
  const float* Wo1  = (const float*)d_in[5];
  const float* b1   = (const float*)d_in[6];
  const float* Wr2  = (const float*)d_in[7];
  const float* Wo2  = (const float*)d_in[8];
  const float* b2   = (const float*)d_in[9];
  const float* Wg   = (const float*)d_in[10];
  const float* bg   = (const float*)d_in[11];
  const float* Wq   = (const float*)d_in[12];
  const float* bq   = (const float*)d_in[13];

  const int N  = in_sizes[0] / GF;      // 2048 nodes
  const int E  = in_sizes[2];           // 65536 edges
  const int BL = in_sizes[3] / GF;      // B*L = 512
  const int L  = 64;
  const int Bq = BL / L;                // 8
  const int Nper = N / Bq;              // 256 nodes per batch item
  const int* srcp = ei;
  const int* dstp = ei + E;

  // ---- workspace arena (256B aligned) ----
  char* wp = (char*)d_ws;
  auto alloc = [&](size_t bytes) -> void* {
    void* r = wp;
    wp += (bytes + 255) & ~(size_t)255;
    return r;
  };
  _Float16* Wcat1 = (_Float16*)alloc((size_t)GF * GK1 * 2);
  _Float16* Wcat2 = (_Float16*)alloc((size_t)GF * GK1 * 2);
  _Float16* WgA16 = (_Float16*)alloc((size_t)GF * GF * 2);
  _Float16* WgB16 = (_Float16*)alloc((size_t)GF * GF * 2);
  _Float16* Wq16  = (_Float16*)alloc((size_t)GF * GK2 * 2);
  _Float16* q16   = (_Float16*)alloc((size_t)BL * GF * 2);
  float*    cnt   = (float*)alloc((size_t)3 * N * 4);
  float*    agg   = (float*)alloc((size_t)3 * N * GF * 4);
  _Float16* Acat  = (_Float16*)alloc((size_t)N * GK1 * 2);
  _Float16* Acat2 = (_Float16*)alloc((size_t)N * GK2 * 2);   // [h16 | qi16]
  float*    hbuf  = (float*)alloc((size_t)N * GF * 4);
  float*    hg    = (float*)alloc((size_t)N * GF * 4);
  float*    qg    = (float*)alloc((size_t)BL * GF * 4);
  float*    qibuf = (float*)alloc((size_t)N * GF * 4);
  float*    zbuf  = (float*)alloc((size_t)N * GF * 4);
  float*    hn1   = (float*)alloc((size_t)N * GF * 4);

  auto cdiv = [](long a, long b) { return (int)((a + b - 1) / b); };
  auto gemm = [&](const _Float16* A, int lda, const _Float16* W, int ldw,
                  const float* bias, float* C, int ldc,
                  _Float16* C16, int ldc16, int M, int Nc, int K, int relu) {
    dim3 g(M / 128, Nc / 64);
    gemm_f16_wmma<<<g, 256, 0, stream>>>(A, lda, W, ldw, bias, C, ldc, C16, ldc16, K, relu);
  };

  // ---- one-time prep ----
  build_wcat<<<cdiv((long)GF * GK1, 256), 256, 0, stream>>>(Wo1, Wr1, Wcat1, (long)GF * GK1);
  build_wcat<<<cdiv((long)GF * GK1, 256), 256, 0, stream>>>(Wo2, Wr2, Wcat2, (long)GF * GK1);
  conv_rows_f16<<<cdiv((long)GF * GF, 256), 256, 0, stream>>>(Wg, GK2, 0,  WgA16, GF,  (long)GF * GF);
  conv_rows_f16<<<cdiv((long)GF * GF, 256), 256, 0, stream>>>(Wg, GK2, GF, WgB16, GF,  (long)GF * GF);
  conv_rows_f16<<<cdiv((long)GF * GK2, 256), 256, 0, stream>>>(Wq, GK2, 0, Wq16, GK2, (long)GF * GK2);
  conv_rows_f16<<<cdiv((long)BL * GF, 256), 256, 0, stream>>>(q, GF, 0, q16, GF, (long)BL * GF);

  zero_f32<<<64, 256, 0, stream>>>(cnt, (long)3 * N);
  count_edges<<<cdiv(E, 256), 256, 0, stream>>>(dstp, et, cnt, E, N);

  // qg = q @ Wg[:,F:]^T + bg    (layer-independent, computed once)
  gemm(q16, GF, WgB16, GF, bg, qg, GF, nullptr, 0, BL, GF, GF, 0);

  // ---- layer = RGCN(relu) + gate ----
  auto layer = [&](const float* x_in, const _Float16* Wcat, const float* bias,
                   float* out_hn) {
    // scatter-mean aggregation per relation
    zero_f32<<<2048, 256, 0, stream>>>(agg, (long)3 * N * GF);
    scatter_edges<<<E, 256, 0, stream>>>(x_in, srcp, dstp, et, agg, N);
    build_acat<<<cdiv((long)N * GK1, 256), 256, 0, stream>>>(x_in, agg, cnt, Acat, N, (long)N * GK1);
    // h = relu(Acat @ Wcat^T + b); h16 -> first half of Acat2
    gemm(Acat, GK1, Wcat, GK1, bias, hbuf, GF, Acat2, GK2, N, GF, GK1, 1);
    // hg = h @ Wg[:,:F]^T
    gemm(Acat2, GK2, WgA16, GF, nullptr, hg, GF, nullptr, 0, N, GF, GF, 0);
    // qi = sum_l sigmoid(hg + qg) * q ; qi16 -> second half of Acat2
    qi_kernel<<<N, 256, 0, stream>>>(hg, qg, q, qibuf, Acat2 + GF, GK2, Nper, L);
    // z = [h|qi] @ Wq^T + bq
    gemm(Acat2, GK2, Wq16, GK2, bq, zbuf, GF, nullptr, 0, N, GF, GK2, 0);
    // hn = sigmoid(z)*tanh(qi) + (1-sigmoid(z))*h
    combine_gate<<<cdiv((long)N * GF, 256), 256, 0, stream>>>(zbuf, qibuf, hbuf, out_hn, (long)N * GF);
  };

  layer(x,   Wcat1, b1, hn1);
  layer(hn1, Wcat2, b2, (float*)d_out);
}
